// SimpleMoE_Optimized_85770496901557
// MI455X (gfx1250) — matrix operand order repaired and checked by reading, not compiled, not run on hardware
//
#include <hip/hip_runtime.h>
#include <stdint.h>

// Problem constants (match reference)
#define H_DIM 1024
#define F_DIM 4096
#define E_NUM 8
#define N_TOK 8192

// GEMM tiling
#define BM 128
#define BN 128
#define BK 64
#define SA 72   // LDS halfword row stride: 144B rows -> 16B-aligned b128, banks spread
#define TILE_HALVES (BM * SA + BN * SA)   // halves per LDS buffer (A then B)
#define BUF_BYTES   (TILE_HALVES * 2)     // bytes per LDS buffer
#define SMEM_BYTES  (2 * BUF_BYTES)       // double buffered

typedef __attribute__((ext_vector_type(16))) __bf16 v16bf;
typedef __attribute__((ext_vector_type(8)))  float  v8f;

__device__ __forceinline__ uint16_t f2bf(float f) {
  union { float f; uint32_t u; } v; v.f = f;
  uint32_t r = v.u + 0x7FFFu + ((v.u >> 16) & 1u);   // round-to-nearest-even
  return (uint16_t)(r >> 16);
}
__device__ __forceinline__ uint32_t f2bf2(float lo, float hi) {
  return (uint32_t)f2bf(lo) | ((uint32_t)f2bf(hi) << 16);
}

// CDNA5 async Global->LDS copy (ASYNCcnt-tracked), 16B per lane.
__device__ __forceinline__ void async_b128(uint32_t lds_off, const uint16_t* g) {
  asm volatile("global_load_async_to_lds_b128 %0, %1, off"
               :: "v"(lds_off), "v"(g) : "memory");
}
__device__ __forceinline__ void wait_async0() {
  asm volatile("s_wait_asynccnt 0x0" ::: "memory");
}

// ---------------------------------------------------------------- zero init
__global__ __launch_bounds__(256)
void moe_zero_kernel(float* __restrict__ out, int* __restrict__ cnt) {
  int i = blockIdx.x * 256 + threadIdx.x;
  const int n4 = (N_TOK * H_DIM) / 4;
  if (i < n4) {
    float4 z; z.x = 0.f; z.y = 0.f; z.z = 0.f; z.w = 0.f;
    ((float4*)out)[i] = z;
  }
  if (blockIdx.x == 0 && threadIdx.x < E_NUM) cnt[threadIdx.x] = 0;
}

// ------------------------------------------------- fp32 -> bf16 (no transpose)
__global__ __launch_bounds__(256)
void convert_x_kernel(const float* __restrict__ in, uint16_t* __restrict__ out) {
  size_t i = (size_t)(blockIdx.x * 256 + threadIdx.x) * 8;
  float4 a = ((const float4*)(in + i))[0];
  float4 b = ((const float4*)(in + i))[1];
  uint4 o;
  o.x = f2bf2(a.x, a.y); o.y = f2bf2(a.z, a.w);
  o.z = f2bf2(b.x, b.y); o.w = f2bf2(b.z, b.w);
  *(uint4*)(out + i) = o;
}

// ------------------------------------------------- fp32 [R][C] -> bf16 [C][R]
__global__ __launch_bounds__(256)
void transpose_bf16_kernel(const float* __restrict__ in, uint16_t* __restrict__ out,
                           int R, int C) {
  __shared__ float T[32][33];
  const size_t slice = (size_t)R * C;
  const float* src = in + (size_t)blockIdx.z * slice;
  uint16_t* dst = out + (size_t)blockIdx.z * slice;
  const int c0 = blockIdx.x * 32, r0 = blockIdx.y * 32;
  const int tid = threadIdx.x;
#pragma unroll
  for (int r = 0; r < 4; ++r) {
    int l = tid + r * 256;
    int rr = l >> 5, cc = l & 31;
    T[rr][cc] = src[(size_t)(r0 + rr) * C + (c0 + cc)];
  }
  __syncthreads();
#pragma unroll
  for (int r = 0; r < 4; ++r) {
    int l = tid + r * 256;
    int oc = l >> 5, orr = l & 31;
    dst[(size_t)(c0 + oc) * R + (r0 + orr)] = f2bf(T[orr][oc]);
  }
}

// ---------------------------------------------------------------- gating
__global__ __launch_bounds__(256)
void moe_gate_kernel(const float* __restrict__ x,
                     const float* __restrict__ gW,
                     const float* __restrict__ gb,
                     int* __restrict__ cnt,
                     int* __restrict__ tok,
                     float* __restrict__ wt) {
  __shared__ float Wg[H_DIM * E_NUM];          // 32 KB
  const int tid = threadIdx.x;
  for (int i = tid; i < H_DIM * E_NUM; i += 256) Wg[i] = gW[i];
  __syncthreads();

  const int t = blockIdx.x * 256 + tid;
  float s[E_NUM];
#pragma unroll
  for (int e = 0; e < E_NUM; ++e) s[e] = gb[e];
  const float* xr = x + (size_t)t * H_DIM;
  for (int h = 0; h < H_DIM; ++h) {
    float xv = xr[h];
    const float4* w4 = (const float4*)&Wg[h * E_NUM];
    float4 wa = w4[0], wb = w4[1];
    s[0] += xv * wa.x; s[1] += xv * wa.y; s[2] += xv * wa.z; s[3] += xv * wa.w;
    s[4] += xv * wb.x; s[5] += xv * wb.y; s[6] += xv * wb.z; s[7] += xv * wb.w;
  }
  int e0 = 0; float s0 = s[0];
#pragma unroll
  for (int e = 1; e < E_NUM; ++e) if (s[e] > s0) { s0 = s[e]; e0 = e; }
  int e1 = (e0 == 0) ? 1 : 0; float s1 = s[e1];
#pragma unroll
  for (int e = 0; e < E_NUM; ++e) if (e != e0 && s[e] > s1) { s1 = s[e]; e1 = e; }
  float w0 = 1.0f / (1.0f + __expf(s1 - s0));   // softmax over {s0,s1}, s0>=s1
  float w1 = 1.0f - w0;
  int p0 = atomicAdd(&cnt[e0], 1);
  tok[e0 * N_TOK + p0] = t;  wt[e0 * N_TOK + p0] = w0;
  int p1 = atomicAdd(&cnt[e1], 1);
  tok[e1 * N_TOK + p1] = t;  wt[e1 * N_TOK + p1] = w1;
}

// ------------------------------------------------- WMMA fragment helpers
// Each fragment half is 16B contiguous and 16B-aligned -> two ds_load_b128.
__device__ __forceinline__ void load_a_frag(v16bf* frag, const uint16_t* lds,
                                            int rowcol, int ks, int half) {
  uint4* p = (uint4*)frag;
  const uint16_t* base = lds + rowcol * SA + ks + 8 * half;
  p[0] = *(const uint4*)(base);        // K pairs {0..7}+8h
  p[1] = *(const uint4*)(base + 16);   // K pairs {16..23}+8h
}
__device__ __forceinline__ void load_b_frag(v16bf* frag, const uint16_t* lds,
                                            int col, int ks, int half) {
  uint4* p = (uint4*)frag;
  const uint16_t* base = lds + col * SA + ks + 8 * half;
  p[0] = *(const uint4*)(base);        // K {0..7}+8h
  p[1] = *(const uint4*)(base + 16);   // K {16..23}+8h
}

// 16 WMMAs on one staged BK=64 LDS tile
__device__ __forceinline__ void wmma_block(const uint16_t* As, const uint16_t* Bs,
                                           v8f (&acc)[2][4], int wm, int wn,
                                           int half, int idx) {
#pragma unroll
  for (int ks = 0; ks < BK; ks += 32) {
    v16bf a[2], b[4];
#pragma unroll
    for (int mi = 0; mi < 2; ++mi)
      load_a_frag(&a[mi], As, wm * 32 + mi * 16 + idx, ks, half);
#pragma unroll
    for (int ni = 0; ni < 4; ++ni)
      load_b_frag(&b[ni], Bs, wn * 64 + ni * 16 + idx, ks, half);
#pragma unroll
    for (int mi = 0; mi < 2; ++mi)
#pragma unroll
      for (int ni = 0; ni < 4; ++ni)
        acc[mi][ni] = __builtin_amdgcn_wmma_f32_16x16x32_bf16(
            false, a[mi], false, b[ni], (short)0, acc[mi][ni], false, false);
  }
}

// ---------------------------------------------------------------- GEMM1
__global__ __launch_bounds__(256)
void moe_ffn1_kernel(const uint16_t* __restrict__ xbf,   // [N_TOK][H_DIM] bf16
                     const uint16_t* __restrict__ W1t,   // [F_DIM][H_DIM] bf16 slice
                     const float* __restrict__ b1,
                     const int* __restrict__ cnt_p,
                     const int* __restrict__ tok,
                     uint16_t* __restrict__ hbuf) {      // [N_TOK][F_DIM] bf16
  extern __shared__ __align__(16) uint16_t smem[];       // 2 x (A tile + B tile)
  const int cnt = *cnt_p;
  const int m_base = blockIdx.y * BM;
  if (m_base >= cnt) return;
  const int n_base = blockIdx.x * BN;

  __shared__ int toks[BM];

  const int tid = threadIdx.x;
  const int lane = tid & 31, wave = tid >> 5;
  const int wm = wave >> 1, wn = wave & 1;
  const int half = lane >> 4, idx = lane & 15;

  if (tid < BM) {
    int r = m_base + tid;
    toks[tid] = tok[(r < cnt) ? r : (cnt - 1)];
  }
  __syncthreads();

  const int rowA = tid >> 3;          // 0..31
  const int ko   = (tid & 7) * 8;     // halfword offset
  const uint16_t* aP[4];
#pragma unroll
  for (int r = 0; r < 4; ++r)
    aP[r] = xbf + (size_t)toks[rowA + r * 32] * H_DIM + ko;
  const uint16_t* bP = W1t + (size_t)(n_base + rowA) * H_DIM + ko;
  uint32_t aL[4], bL[4];              // LDS byte offsets (buffer 0)
#pragma unroll
  for (int r = 0; r < 4; ++r) {
    aL[r] = (uint32_t)(uintptr_t)&smem[(rowA + r * 32) * SA + ko];
    bL[r] = (uint32_t)(uintptr_t)&smem[BM * SA + (rowA + r * 32) * SA + ko];
  }

  v8f acc[2][4];
#pragma unroll
  for (int mi = 0; mi < 2; ++mi)
#pragma unroll
    for (int ni = 0; ni < 4; ++ni)
#pragma unroll
      for (int j = 0; j < 8; ++j) acc[mi][ni][j] = 0.0f;

  // Compile-time-constant buffer pointers
  const uint16_t* A0 = smem;
  const uint16_t* B0 = smem + BM * SA;
  const uint16_t* A1 = smem + TILE_HALVES;
  const uint16_t* B1 = A1 + BM * SA;

  // Prologue: async-stage tile 0 into buffer 0
#pragma unroll
  for (int r = 0; r < 4; ++r) async_b128(aL[r], aP[r]);
#pragma unroll
  for (int r = 0; r < 4; ++r) async_b128(bL[r], bP + r * (32 * H_DIM));

  const int nT = H_DIM / BK;          // 16 tiles (even)
  for (int t = 0; t < nT; t += 2) {
    // phase A: compute tile t (buf0), stage tile t+1 -> buf1
    wait_async0();
    __syncthreads();
    {
      int kk = (t + 1) * BK;
#pragma unroll
      for (int r = 0; r < 4; ++r) async_b128(aL[r] + BUF_BYTES, aP[r] + kk);
#pragma unroll
      for (int r = 0; r < 4; ++r)
        async_b128(bL[r] + BUF_BYTES, bP + kk + r * (32 * H_DIM));
      __builtin_prefetch(bP + kk + BK, 0, 1);
    }
    wmma_block(A0, B0, acc, wm, wn, half, idx);
    // phase B: compute tile t+1 (buf1), stage tile t+2 -> buf0
    wait_async0();
    __syncthreads();
    if (t + 2 < nT) {
      int kk = (t + 2) * BK;
#pragma unroll
      for (int r = 0; r < 4; ++r) async_b128(aL[r], aP[r] + kk);
#pragma unroll
      for (int r = 0; r < 4; ++r)
        async_b128(bL[r], bP + kk + r * (32 * H_DIM));
    }
    wmma_block(A1, B1, acc, wm, wn, half, idx);
  }

#pragma unroll
  for (int mi = 0; mi < 2; ++mi)
#pragma unroll
    for (int ni = 0; ni < 4; ++ni) {
      int ng = n_base + wn * 64 + ni * 16 + idx;
      float bias = b1[ng];
#pragma unroll
      for (int r = 0; r < 8; ++r) {
        int lm = wm * 32 + mi * 16 + half * 8 + r;   // C/D layout: VGPR r, lane-half
        int row = m_base + lm;
        if (row < cnt) {
          float v = acc[mi][ni][r] + bias;
          v = fmaxf(v, 0.0f);
          hbuf[(size_t)row * F_DIM + ng] = f2bf(v);
        }
      }
    }
}

// ---------------------------------------------------------------- GEMM2
__global__ __launch_bounds__(256)
void moe_ffn2_kernel(const uint16_t* __restrict__ hbuf,  // [N_TOK][F_DIM] bf16
                     const uint16_t* __restrict__ W2t,   // [H_DIM][F_DIM] bf16 slice
                     const float* __restrict__ b2,
                     const int* __restrict__ cnt_p,
                     const int* __restrict__ tok,
                     const float* __restrict__ wt,
                     float* __restrict__ out) {
  extern __shared__ __align__(16) uint16_t smem[];
  const int cnt = *cnt_p;
  const int m_base = blockIdx.y * BM;
  if (m_base >= cnt) return;
  const int n_base = blockIdx.x * BN;

  __shared__ int toks[BM];
  __shared__ float wts[BM];

  const int tid = threadIdx.x;
  const int lane = tid & 31, wave = tid >> 5;
  const int wm = wave >> 1, wn = wave & 1;
  const int half = lane >> 4, idx = lane & 15;

  if (tid < BM) {
    int r = m_base + tid;
    bool v = r < cnt;
    toks[tid] = v ? tok[r] : 0;
    wts[tid]  = v ? wt[r] : 0.0f;
  }
  __syncthreads();

  const int rowA = tid >> 3;
  const int ko   = (tid & 7) * 8;
  const uint16_t* aP = hbuf + (size_t)(m_base + rowA) * F_DIM + ko;
  const uint16_t* bP = W2t + (size_t)(n_base + rowA) * F_DIM + ko;
  uint32_t aL[4], bL[4];
#pragma unroll
  for (int r = 0; r < 4; ++r) {
    aL[r] = (uint32_t)(uintptr_t)&smem[(rowA + r * 32) * SA + ko];
    bL[r] = (uint32_t)(uintptr_t)&smem[BM * SA + (rowA + r * 32) * SA + ko];
  }

  v8f acc[2][4];
#pragma unroll
  for (int mi = 0; mi < 2; ++mi)
#pragma unroll
    for (int ni = 0; ni < 4; ++ni)
#pragma unroll
      for (int j = 0; j < 8; ++j) acc[mi][ni][j] = 0.0f;

  const uint16_t* A0 = smem;
  const uint16_t* B0 = smem + BM * SA;
  const uint16_t* A1 = smem + TILE_HALVES;
  const uint16_t* B1 = A1 + BM * SA;

#pragma unroll
  for (int r = 0; r < 4; ++r) async_b128(aL[r], aP + r * (32 * F_DIM));
#pragma unroll
  for (int r = 0; r < 4; ++r) async_b128(bL[r], bP + r * (32 * F_DIM));

  const int nT = F_DIM / BK;          // 64 tiles (even)
  for (int t = 0; t < nT; t += 2) {
    wait_async0();
    __syncthreads();
    {
      int kk = (t + 1) * BK;
#pragma unroll
      for (int r = 0; r < 4; ++r)
        async_b128(aL[r] + BUF_BYTES, aP + kk + r * (32 * F_DIM));
#pragma unroll
      for (int r = 0; r < 4; ++r)
        async_b128(bL[r] + BUF_BYTES, bP + kk + r * (32 * F_DIM));
      __builtin_prefetch(bP + kk + BK, 0, 1);
    }
    wmma_block(A0, B0, acc, wm, wn, half, idx);
    wait_async0();
    __syncthreads();
    if (t + 2 < nT) {
      int kk = (t + 2) * BK;
#pragma unroll
      for (int r = 0; r < 4; ++r)
        async_b128(aL[r], aP + kk + r * (32 * F_DIM));
#pragma unroll
      for (int r = 0; r < 4; ++r)
        async_b128(bL[r], bP + kk + r * (32 * F_DIM));
    }
    wmma_block(A1, B1, acc, wm, wn, half, idx);
  }

#pragma unroll
  for (int mi = 0; mi < 2; ++mi)
#pragma unroll
    for (int ni = 0; ni < 4; ++ni) {
      int ng = n_base + wn * 64 + ni * 16 + idx;
      float bias = b2[ng];
#pragma unroll
      for (int r = 0; r < 8; ++r) {
        int lm = wm * 32 + mi * 16 + half * 8 + r;
        if (m_base + lm < cnt) {
          float yv = acc[mi][ni][r] + bias;
          float* dst = &out[(size_t)toks[lm] * H_DIM + ng];
          __hip_atomic_fetch_add(dst, wts[lm] * yv,
                                 __ATOMIC_RELAXED, __HIP_MEMORY_SCOPE_AGENT);
        }
      }
    }
}

// ---------------------------------------------------------------- launcher
extern "C" void kernel_launch(void* const* d_in, const int* in_sizes, int n_in,
                              void* d_out, int out_size, void* d_ws, size_t ws_size,
                              hipStream_t stream) {
  (void)in_sizes; (void)n_in; (void)out_size; (void)ws_size;
  const float* x   = (const float*)d_in[0];
  const float* gW  = (const float*)d_in[1];
  const float* gb  = (const float*)d_in[2];
  const float* W1  = (const float*)d_in[3];
  const float* b1  = (const float*)d_in[4];
  const float* W2  = (const float*)d_in[5];
  const float* b2  = (const float*)d_in[6];
  float* out = (float*)d_out;

  // Workspace layout (bytes)
  char* ws = (char*)d_ws;
  const size_t SZ_TOK = (size_t)E_NUM * N_TOK * 4;          // 256 KB
  const size_t SZ_XBF = (size_t)N_TOK * H_DIM * 2;          // 16 MB
  const size_t SZ_W   = (size_t)E_NUM * H_DIM * F_DIM * 2;  // 64 MB each
  int*      cnt  = (int*)ws;
  int*      tok  = (int*)(ws + 256);
  float*    wtl  = (float*)(ws + 256 + SZ_TOK);
  uint16_t* xbf  = (uint16_t*)(ws + 256 + 2 * SZ_TOK);
  uint16_t* W1t  = (uint16_t*)(ws + 256 + 2 * SZ_TOK + SZ_XBF);
  uint16_t* W2t  = (uint16_t*)(ws + 256 + 2 * SZ_TOK + SZ_XBF + SZ_W);
  uint16_t* hbuf = (uint16_t*)(ws + 256 + 2 * SZ_TOK + SZ_XBF + 2 * SZ_W);

  const int zero_blocks = (N_TOK * H_DIM / 4 + 255) / 256;
  moe_zero_kernel<<<zero_blocks, 256, 0, stream>>>(out, cnt);
  convert_x_kernel<<<N_TOK * H_DIM / 8 / 256, 256, 0, stream>>>(x, xbf);
  dim3 tg1(F_DIM / 32, H_DIM / 32, E_NUM);
  transpose_bf16_kernel<<<tg1, 256, 0, stream>>>(W1, W1t, H_DIM, F_DIM);
  dim3 tg2(H_DIM / 32, F_DIM / 32, E_NUM);
  transpose_bf16_kernel<<<tg2, 256, 0, stream>>>(W2, W2t, F_DIM, H_DIM);

  moe_gate_kernel<<<N_TOK / 256, 256, 0, stream>>>(x, gW, gb, cnt, tok, wtl);

  dim3 g1(F_DIM / BN, N_TOK / BM);   // (32, 64) — M blocks self-trim via cnt
  dim3 g2(H_DIM / BN, N_TOK / BM);   // (8, 64)
  for (int e = 0; e < E_NUM; ++e) {
    moe_ffn1_kernel<<<g1, 256, SMEM_BYTES, stream>>>(
        xbf, W1t + (size_t)e * H_DIM * F_DIM, b1 + (size_t)e * F_DIM,
        cnt + e, tok + (size_t)e * N_TOK, hbuf);
    moe_ffn2_kernel<<<g2, 256, SMEM_BYTES, stream>>>(
        hbuf, W2t + (size_t)e * F_DIM * H_DIM, b2 + (size_t)e * H_DIM,
        cnt + e, tok + (size_t)e * N_TOK, wtl + (size_t)e * N_TOK, out);
  }
}
